// MultiHeadAttention_85409719648800
// MI455X (gfx1250) — matrix operand order repaired and checked by reading, not compiled, not run on hardware
//
#include <hip/hip_runtime.h>
#include <math.h>

typedef __attribute__((ext_vector_type(16))) _Float16 v16h;
typedef __attribute__((ext_vector_type(8)))  float    v8f;

#define B_   4
#define S_   2048
#define DIN  512
#define H_   8
#define DO_  64

// ---------------- CDNA5 helpers ----------------

// LDS byte-offset of a generic pointer that points into __shared__ memory.
static __device__ __forceinline__ unsigned lds_bytes(void* p) {
  return (unsigned)(unsigned long long)(__attribute__((address_space(3))) void*)p;
}

// Async copy 16B/lane global -> LDS (tracked by ASYNCcnt).
static __device__ __forceinline__ void async_cp_b128(void* ldsDst, const void* gSrc) {
  asm volatile("global_load_async_to_lds_b128 %0, %1, off"
               :: "v"(lds_bytes(ldsDst)), "v"(gSrc)
               : "memory");
}

#define WAIT_ASYNC(n) asm volatile("s_wait_asynccnt " #n ::: "memory")

// ---------- WMMA fragment helpers (ISA 7.12.2 layouts, wave32) ----------

// A fragment: 16x32 f16, row-major source with leading dim lda (contiguous 8-half runs).
static __device__ __forceinline__ v16h load_a_h(const _Float16* src, int lda) {
  const int lane = threadIdx.x & 31;
  const int m  = lane & 15;
  const int kb = (lane & 16) ? 8 : 0;
  const _Float16* row = src + m * lda + kb;
  v16h a;
#pragma unroll
  for (int i = 0; i < 8; ++i) a[i] = row[i];
#pragma unroll
  for (int i = 0; i < 8; ++i) a[8 + i] = row[16 + i];
  return a;
}

static __device__ __forceinline__ v16h load_a_f(const float* src, int lda) {
  const int lane = threadIdx.x & 31;
  const int m  = lane & 15;
  const int kb = (lane & 16) ? 8 : 0;
  const float* row = src + m * lda + kb;
  v16h a;
#pragma unroll
  for (int i = 0; i < 8; ++i) a[i] = (_Float16)row[i];
#pragma unroll
  for (int i = 0; i < 8; ++i) a[8 + i] = (_Float16)row[16 + i];
  return a;
}

// B fragment from a COLUMN-of-B-major f16 tile: tile[colN][ldr]; each lane reads
// K-rows kb..kb+7 and kb+16..kb+23 of its column (kb = 0 or 8) -> 2x b128 loads.
// Works for both LDS tiles and global transposed weight tiles.
static __device__ __forceinline__ v16h load_b_cm(const _Float16* tile, int ldr,
                                                 int coln, int kk) {
  const int lane = threadIdx.x & 31;
  const int n  = lane & 15;
  const int kb = (lane & 16) ? 8 : 0;
  const _Float16* p = tile + (coln + n) * ldr + kk + kb;
  v16h b;
#pragma unroll
  for (int i = 0; i < 8; ++i) b[i] = p[i];
#pragma unroll
  for (int i = 0; i < 8; ++i) b[8 + i] = p[16 + i];
  return b;
}

static __device__ __forceinline__ v8f wmma32(v16h a, v16h b, v8f c) {
  return __builtin_amdgcn_wmma_f32_16x16x32_f16(false, a, false, b, (short)0, c, false, false);
}

// C/D layout: col n = lane&15 ; VGPR r holds row r (lanes 0-15) or row r+8 (lanes 16-31)

// ---------- Kernel 0: convert + transpose weights to f16 ----------
// w  [H,3,DIN,DO] f32  ->  w16t  [H*3][DO][DIN] f16  (column-of-B-major)
// wo [DIN,DO]     f32  ->  wo16t [DO][DIN]      f16
__global__ void __launch_bounds__(256)
weight_prep_kernel(const float* __restrict__ w, const float* __restrict__ wo,
                   _Float16* __restrict__ w16t, _Float16* __restrict__ wo16t) {
  const int idx = blockIdx.x * 256 + threadIdx.x;
  const int NW = H_ * 3 * DIN * DO_;              // 786432
  if (idx < NW) {
    const int hc = idx / (DO_ * DIN);
    const int rem = idx - hc * (DO_ * DIN);
    const int o = rem / DIN;
    const int i = rem - o * DIN;
    w16t[idx] = (_Float16)w[(hc * DIN + i) * DO_ + o];
  } else {
    const int j = idx - NW;                       // 0 .. DIN*DO_-1
    if (j < DIN * DO_) {
      const int o = j / DIN;
      const int i = j - o * DIN;
      wo16t[j] = (_Float16)wo[i * DO_ + o];
    }
  }
}

// ---------- Kernel 1: QKV projections ----------
// one wave = one 16x16 tile of one (h, c, ntile).
// Q, K stored [B,H,S,DO]; V stored transposed [B,H,DO,S].
__global__ void __launch_bounds__(256)
qkv_proj_kernel(const float* __restrict__ x, const _Float16* __restrict__ w16t,
                _Float16* __restrict__ Q, _Float16* __restrict__ K,
                _Float16* __restrict__ Vt) {
  int job = __builtin_amdgcn_readfirstlane(blockIdx.x * 8 + (threadIdx.x >> 5));
  const int nt = job & 3; job >>= 2;
  const int c  = job % 3; job /= 3;
  const int h  = job & 7; job >>= 3;
  const int rt = job;                     // 0..511  (B*S/16 row tiles)

  v8f acc = {};
  const float* xbase = x + rt * 16 * DIN;
  const _Float16* wbase = w16t + (h * 3 + c) * DO_ * DIN;   // [DO][DIN]
  for (int k = 0; k < DIN; k += 32) {
    v16h a = load_a_f(xbase + k, DIN);
    v16h b = load_b_cm(wbase, DIN, nt * 16, k);
    acc = wmma32(a, b, acc);
  }

  const int lane = threadIdx.x & 31;
  const int n  = lane & 15;
  const int hi = (lane >> 4) & 1;
#pragma unroll
  for (int r = 0; r < 8; ++r) {
    const int m   = r + hi * 8;
    const int tok = rt * 16 + m;          // global token 0..8191
    const int bb  = tok >> 11;            // / S_
    const int s   = tok & (S_ - 1);
    const _Float16 val = (_Float16)acc[r];
    if (c == 0)                           // wave-uniform (scalar) branch
      Q[((bb * H_ + h) * S_ + s) * DO_ + nt * 16 + n] = val;
    else if (c == 1)
      K[((bb * H_ + h) * S_ + s) * DO_ + nt * 16 + n] = val;
    else
      Vt[((bb * H_ + h) * DO_ + nt * 16 + n) * S_ + s] = val;
  }
}

// ---------- Kernel 2: fused flash attention ----------
// Block = 8 waves sharing one (b,h); wave w owns one q-tile. K/V tiles staged
// into LDS with double-buffered global_load_async_to_lds_b128, shared 8x.
__global__ void __launch_bounds__(256)
attn_kernel(const _Float16* __restrict__ Q, const _Float16* __restrict__ K,
            const _Float16* __restrict__ Vt, _Float16* __restrict__ O) {
  __shared__ _Float16 ktile[2][32][DO_];  // [key][d]   4KB per buffer
  __shared__ _Float16 vtile[2][DO_][32];  // [d][key]   4KB per buffer
  __shared__ _Float16 plds[8][16 * 32];   // per-wave P-transpose staging

  const int wid = __builtin_amdgcn_readfirstlane(threadIdx.x >> 5);
  int job = __builtin_amdgcn_readfirstlane(blockIdx.x * 8 + wid);
  const int qt = job & 127; job >>= 7;    // h, bb are block-uniform (8-aligned)
  const int h  = job & 7;   job >>= 3;
  const int bb = job;

  const int lane = threadIdx.x & 31;
  const int n  = lane & 15;
  const int hi = (lane >> 4) & 1;

  const _Float16* Qb  = Q  + ((bb * H_ + h) * S_ + qt * 16) * DO_;
  const _Float16* Kb  = K  + (bb * H_ + h) * S_ * DO_;
  const _Float16* Vtb = Vt + (bb * H_ + h) * DO_ * S_;

  // per-thread async-copy coordinates (16B chunks, coalesced)
  const int tid  = threadIdx.x;
  const int krow = tid >> 3, kseg = tid & 7;   // K tile: 32 rows x 128B
  const int vrow = tid >> 2, vseg = tid & 3;   // V tile: 64 rows x 64B

  const v16h qa0 = load_a_h(Qb, DO_);       // Q k = 0..31
  const v16h qa1 = load_a_h(Qb + 32, DO_);  // Q k = 32..63

  float mrun[8], lrun[8];
#pragma unroll
  for (int r = 0; r < 8; ++r) { mrun[r] = -INFINITY; lrun[r] = 0.0f; }
  v8f o0 = {}, o1 = {}, o2 = {}, o3 = {};

  // stage tile j into buffer buf (2 async instructions per wave)
  auto stage = [&](int j, int buf) {
    async_cp_b128(&ktile[buf][krow][kseg * 8], Kb + (j * 32 + krow) * DO_ + kseg * 8);
    async_cp_b128(&vtile[buf][vrow][vseg * 8], Vtb + vrow * S_ + j * 32 + vseg * 8);
  };

  constexpr int NJ = S_ / 32;
  stage(0, 0);

  for (int j = 0; j < NJ; ++j) {
    const int cur = j & 1;
    if (j + 1 < NJ) {
      stage(j + 1, cur ^ 1);
      WAIT_ASYNC(2);                      // tile j complete (in-order), j+1 in flight
    } else {
      WAIT_ASYNC(0);
    }
    __syncthreads();                      // tile j visible to all waves

    const _Float16* kt = &ktile[cur][0][0];
    const _Float16* vt = &vtile[cur][0][0];

    // ---- scores: two 16x16 tiles covering 32 keys ----
    v8f s0 = {}, s1 = {};
    s0 = wmma32(qa0, load_b_cm(kt, DO_, 0, 0), s0);
    s0 = wmma32(qa1, load_b_cm(kt, DO_, 0, 32), s0);
    s1 = wmma32(qa0, load_b_cm(kt, DO_, 16, 0), s1);
    s1 = wmma32(qa1, load_b_cm(kt, DO_, 16, 32), s1);

    // ---- online softmax ----
    float newm[8], alpha[8];
#pragma unroll
    for (int r = 0; r < 8; ++r) {
      const float v0 = s0[r] * 0.125f;    // 1/sqrt(64)
      const float v1 = s1[r] * 0.125f;
      s0[r] = v0; s1[r] = v1;
      float t = fmaxf(v0, v1);
#pragma unroll
      for (int msk = 1; msk <= 8; msk <<= 1)  // reduce within 16-lane half
        t = fmaxf(t, __shfl_xor(t, msk, 32));
      newm[r]  = fmaxf(mrun[r], t);
      alpha[r] = __expf(mrun[r] - newm[r]);
      mrun[r]  = newm[r];
    }
#pragma unroll
    for (int r = 0; r < 8; ++r) {
      const float p0 = __expf(s0[r] - newm[r]);
      const float p1 = __expf(s1[r] - newm[r]);
      float rs = p0 + p1;
#pragma unroll
      for (int msk = 1; msk <= 8; msk <<= 1)
        rs += __shfl_xor(rs, msk, 32);
      lrun[r] = lrun[r] * alpha[r] + rs;
      const int m = r + hi * 8;
      plds[wid][m * 32 + n]      = (_Float16)p0;   // transpose P via LDS
      plds[wid][m * 32 + 16 + n] = (_Float16)p1;
      o0[r] *= alpha[r]; o1[r] *= alpha[r];
      o2[r] *= alpha[r]; o3[r] *= alpha[r];
    }

    // ---- O += P (16x32) x V (32x64) ----
    const v16h pa = load_a_h(&plds[wid][0], 32);   // same-wave LDS, in-order
    o0 = wmma32(pa, load_b_cm(vt, 32, 0,  0), o0);
    o1 = wmma32(pa, load_b_cm(vt, 32, 16, 0), o1);
    o2 = wmma32(pa, load_b_cm(vt, 32, 32, 0), o2);
    o3 = wmma32(pa, load_b_cm(vt, 32, 48, 0), o3);

    __syncthreads();                      // done reading tile j before overwrite
  }

  // ---- epilogue: normalize, concat heads into O[B,S,H*DO] ----
#pragma unroll
  for (int r = 0; r < 8; ++r) {
    const float inv = 1.0f / lrun[r];
    const int m = r + hi * 8;
    const int srow = qt * 16 + m;
    _Float16* ob = O + ((size_t)(bb * S_ + srow)) * (H_ * DO_) + h * DO_;
    ob[0  + n] = (_Float16)(o0[r] * inv);
    ob[16 + n] = (_Float16)(o1[r] * inv);
    ob[32 + n] = (_Float16)(o2[r] * inv);
    ob[48 + n] = (_Float16)(o3[r] * inv);
  }
}

// ---------- Kernel 3: output projection ----------
__global__ void __launch_bounds__(256)
out_proj_kernel(const _Float16* __restrict__ O, const _Float16* __restrict__ wo16t,
                float* __restrict__ out) {
  int job = __builtin_amdgcn_readfirstlane(blockIdx.x * 8 + (threadIdx.x >> 5));
  const int nt = job & 3; job >>= 2;
  const int rt = job;

  v8f acc = {};
  const _Float16* ob = O + (size_t)rt * 16 * (H_ * DO_);
  for (int k = 0; k < H_ * DO_; k += 32) {
    v16h a = load_a_h(ob + k, H_ * DO_);
    v16h b = load_b_cm(wo16t, H_ * DO_, nt * 16, k);
    acc = wmma32(a, b, acc);
  }

  const int lane = threadIdx.x & 31;
  const int n  = lane & 15;
  const int hi = (lane >> 4) & 1;
#pragma unroll
  for (int r = 0; r < 8; ++r) {
    const int m = r + hi * 8;
    out[(rt * 16 + m) * DO_ + nt * 16 + n] = acc[r];
  }
}

// ---------- launcher ----------
extern "C" void kernel_launch(void* const* d_in, const int* in_sizes, int n_in,
                              void* d_out, int out_size, void* d_ws, size_t ws_size,
                              hipStream_t stream) {
  (void)in_sizes; (void)n_in; (void)out_size; (void)ws_size;
  const float* x  = (const float*)d_in[0];   // [B,S,DIN]
  const float* w  = (const float*)d_in[1];   // [H,3,DIN,DO]
  const float* wo = (const float*)d_in[2];   // [H*DO, DO]
  float* out = (float*)d_out;                // [B,S,DO]

  char* ws = (char*)d_ws;
  const size_t qkvBytes = (size_t)B_ * H_ * S_ * DO_ * sizeof(_Float16); // 8 MiB
  _Float16* Q    = (_Float16*)(ws);
  _Float16* K    = (_Float16*)(ws + qkvBytes);
  _Float16* Vt   = (_Float16*)(ws + 2 * qkvBytes);
  _Float16* O    = (_Float16*)(ws + 3 * qkvBytes);
  _Float16* w16t = (_Float16*)(ws + 4 * qkvBytes);
  _Float16* wo16t= (_Float16*)(ws + 4 * qkvBytes
                               + (size_t)H_ * 3 * DIN * DO_ * sizeof(_Float16));

  // convert + transpose weights: 786432 + 32768 elements
  weight_prep_kernel<<<3200, 256, 0, stream>>>(w, wo, w16t, wo16t);
  // 512 row-tiles * 8 heads * 3 (q,k,v) * 4 n-tiles = 49152 wave-jobs, 8 waves/block
  qkv_proj_kernel<<<6144, 256, 0, stream>>>(x, w16t, Q, K, Vt);
  // 4 * 8 * 128 q-tiles = 4096 wave-jobs, 8 waves/block share (b,h) + LDS tiles
  attn_kernel<<<512, 256, 0, stream>>>(Q, K, Vt, O);
  // 512 row-tiles * 4 n-tiles = 2048 wave-jobs
  out_proj_kernel<<<256, 256, 0, stream>>>(O, wo16t, out);
}
